// BertCRF_50354196579075
// MI455X (gfx1250) — matrix operand order repaired and checked by reading, not compiled, not run on hardware
//
#include <hip/hip_runtime.h>
#include <hip/hip_bf16.h>

// ---- problem constants (from reference) ----
constexpr int Bb = 256;          // batch
constexpr int Ls = 512;          // sequence length
constexpr int Hh = 768;          // hidden
constexpr int Kk = 64;           // tags / CRF states
constexpr int HT = Hh / 32;      // 24 k-tiles of 32 for bf16 WMMA
constexpr int NT = Kk / 16;      // 4 n-tiles of 16

#define LOG2E 1.4426950408889634f
#define LN2   0.6931471805599453f

typedef __attribute__((ext_vector_type(16))) __bf16 v16bf;
typedef __attribute__((ext_vector_type(8)))  float  v8f;

// ---------------------------------------------------------------------------
// Kernel 1: pack W (H x K, f32 row-major) into the exact bf16 B-operand
// layout for v_wmma_f32_16x16x32_bf16:
//   per (n-tile, h-tile): 32 lanes x 16 bf16, lane<16 -> col n0+lane, K=0..7,16..23
//                                            lane>=16 -> col n0+lane-16, K=8..15,24..31
// so each lane's B fragment is one contiguous 32-byte chunk.
// ---------------------------------------------------------------------------
__global__ void packW(const float* __restrict__ W, __bf16* __restrict__ Wp) {
  int idx = blockIdx.x * blockDim.x + threadIdx.x;
  if (idx >= NT * HT * 32 * 16) return;
  int e    = idx & 15;
  int lane = (idx >> 4) & 31;
  int rem  = idx >> 9;
  int ht   = rem % HT;
  int nt   = rem / HT;
  int n = nt * 16 + (lane & 15);
  int k = ((lane >> 4) << 3) + (e & 7) + ((e >> 3) << 4);      // lane<16: k=0..7,16..23 ; lane>=16: k=8..15,24..31
  int h = ht * 32 + k;
  Wp[idx] = (__bf16)W[h * Kk + n];
}

// ---------------------------------------------------------------------------
// Kernel 2: emission GEMM  emit[m, n] = sum_h F[m,h] * W[h,n] + bias[n]
// One wave per 16x16 output tile; 4 waves per block share the same 16 rows
// (A loads hit L0). 24 iterations of v_wmma_f32_16x16x32_bf16.
// Memory-bound: 402 MB feature reads @ 23.3 TB/s ~= 19 us floor.
// ---------------------------------------------------------------------------
__global__ void __launch_bounds__(128) emitGemm(const float* __restrict__ F,
                                                const __bf16* __restrict__ Wp,
                                                const float* __restrict__ bias,
                                                float* __restrict__ emit) {
  const int lane = threadIdx.x & 31;
  const int wv   = threadIdx.x >> 5;        // n-tile index 0..3
  const int m0   = blockIdx.x * 16;
  const int row  = lane & 15;               // A row (M) / C column (N)
  const int hi   = lane >> 4;               // half-wave selector
  const float*  arow = F + (size_t)(m0 + row) * Hh + hi * 8;
  const __bf16* wpw  = Wp + (size_t)wv * HT * 32 * 16;

  v8f acc = {};
  for (int ht = 0; ht < HT; ++ht) {
    const float* a0 = arow + ht * 32;
    // A fragment: lanes 0-15 carry K = h0+0..7 and h0+16..23 of row M=lane;
    //             lanes 16-31 carry K = h0+8..15 and h0+24..31 of row M=lane-16.
    float4 f0 = *(const float4*)(a0);
    float4 f1 = *(const float4*)(a0 + 4);
    float4 f2 = *(const float4*)(a0 + 16);
    float4 f3 = *(const float4*)(a0 + 20);
    v16bf a;
    a[0]=(__bf16)f0.x;  a[1]=(__bf16)f0.y;  a[2]=(__bf16)f0.z;  a[3]=(__bf16)f0.w;
    a[4]=(__bf16)f1.x;  a[5]=(__bf16)f1.y;  a[6]=(__bf16)f1.z;  a[7]=(__bf16)f1.w;
    a[8]=(__bf16)f2.x;  a[9]=(__bf16)f2.y;  a[10]=(__bf16)f2.z; a[11]=(__bf16)f2.w;
    a[12]=(__bf16)f3.x; a[13]=(__bf16)f3.y; a[14]=(__bf16)f3.z; a[15]=(__bf16)f3.w;

    v16bf bm = *(const v16bf*)(wpw + ((size_t)(ht * 32 + lane)) * 16);

    acc = __builtin_amdgcn_wmma_f32_16x16x32_bf16(
        /*neg_a=*/false, a, /*neg_b=*/false, bm,
        /*c_mod=*/(short)0, acc, /*reuse_a=*/false, /*reuse_b=*/false);
  }

  // C/D layout: lanes 0-15: VGPR v -> (M=v, N=lane); lanes 16-31: (M=v+8, N=lane-16)
  const int n0 = wv * 16;
  const float bv = bias[n0 + row];
  float* orow = emit + (size_t)(m0 + hi * 8) * Kk + n0 + row;
#pragma unroll
  for (int v = 0; v < 8; ++v)
    orow[(size_t)v * Kk] = acc[v] + bv;
}

// ---------------------------------------------------------------------------
// Kernel 3: CRF forward scan + gold-path score. One block (64 threads = 2
// waves) per batch element; thread j owns state j. T[:,j]*log2e lives in 64
// VGPRs for the whole scan (loop-invariant); d broadcast via ping-pong LDS
// buffers so only ONE barrier per timestep sits on the sequential critical
// path. Next step's emit/mask are prefetched before the barrier so their
// latency overlaps the barrier wait. All math in log2 domain (raw v_exp_f32 /
// v_log_f32, no per-element scale muls); convert with ln2 once at the end.
// ---------------------------------------------------------------------------
__global__ void __launch_bounds__(64) crfScan(const float* __restrict__ emit,
                                              const float* __restrict__ trans,
                                              const int*   __restrict__ tags,
                                              const int*   __restrict__ mask,
                                              float* __restrict__ out) {
  const int j = threadIdx.x;
  const int b = blockIdx.x;
  __shared__ float dsbuf[2][Kk];
  __shared__ float red[Kk];

  const float* eb = emit + (size_t)b * Ls * Kk;
  const int*   tb = tags + (size_t)b * Ls;
  const int*   mb = mask + (size_t)b * Ls;

  // transition column for this state, pre-scaled to log2 domain (loop-invariant)
  float tcol[Kk];
#pragma unroll
  for (int i = 0; i < Kk; ++i) tcol[i] = trans[i * Kk + j] * LOG2E;

  // gold path score partial: thread j handles timesteps j, j+64, ...
  float g = 0.f;
  for (int t = j; t < Ls; t += Kk) {
    if (mb[t]) {
      float sc = eb[t * Kk + tb[t]];
      if (t > 0) sc += trans[tb[t - 1] * Kk + tb[t]];
      g += sc;
    }
  }
  red[j] = g;

  float d = eb[j] * LOG2E;     // d0 = emit[b,0,:]
  dsbuf[0][j] = d;
  __syncthreads();

  // operands for step t=1 already in flight
  float ej = eb[Kk + j];
  int   mt = mb[1];
  int   cur = 0;

  for (int t = 1; t < Ls; ++t) {
    const float* dsc = dsbuf[cur];
    float x[Kk];
    float m = -1e30f;
#pragma unroll
    for (int i = 0; i < Kk; ++i) {
      x[i] = dsc[i] + tcol[i];
      m = fmaxf(m, x[i]);
    }
    // prefetch next step's operands (overlaps with exp burst + barrier)
    float ejn = 0.f;
    int   mtn = 0;
    if (t + 1 < Ls) { ejn = eb[(t + 1) * Kk + j]; mtn = mb[t + 1]; }

    float s = 0.f;
#pragma unroll
    for (int i = 0; i < Kk; ++i)
      s += __builtin_amdgcn_exp2f(x[i] - m);   // raw v_exp_f32 (base-2 domain)
    const float dn = m + __builtin_amdgcn_logf(s) + ej * LOG2E;
    if (mt) d = dn;

    dsbuf[cur ^ 1][j] = d;
    __syncthreads();          // single barrier: publishes new d AND guarantees
                              // all reads of dsbuf[cur] are done before step
                              // t+1 overwrites it
    cur ^= 1;
    ej = ejn;
    mt = mtn;
  }

  // log_z = ln2 * log2sumexp2(d')  (computed redundantly by every thread)
  const float* dsf = dsbuf[cur];
  float m2 = -1e30f;
#pragma unroll
  for (int i = 0; i < Kk; ++i) m2 = fmaxf(m2, dsf[i]);
  float s2 = 0.f;
#pragma unroll
  for (int i = 0; i < Kk; ++i) s2 += __builtin_amdgcn_exp2f(dsf[i] - m2);
  const float logz = (m2 + __builtin_amdgcn_logf(s2)) * LN2;

  if (j == 0) {
    float tot = 0.f;
    for (int i = 0; i < Kk; ++i) tot += red[i];
    out[b] = logz - tot;
  }
}

// ---------------------------------------------------------------------------
extern "C" void kernel_launch(void* const* d_in, const int* in_sizes, int n_in,
                              void* d_out, int out_size, void* d_ws, size_t ws_size,
                              hipStream_t stream) {
  (void)in_sizes; (void)n_in; (void)out_size; (void)ws_size;

  const float* F     = (const float*)d_in[0];   // features (B,L,H) f32
  const float* W     = (const float*)d_in[1];   // W (H,K) f32
  const float* bias  = (const float*)d_in[2];   // b (K,) f32
  const float* trans = (const float*)d_in[3];   // transition (K,K) f32
  const int*   tags  = (const int*)d_in[4];     // tags (B,L) int
  const int*   mask  = (const int*)d_in[5];     // mask (B,L) int/bool
  float*       out   = (float*)d_out;           // (B,) f32

  char* ws = (char*)d_ws;
  float*  emit = (float*)ws;                                    // 32 MB
  __bf16* Wp   = (__bf16*)(ws + (size_t)Bb * Ls * Kk * sizeof(float)); // 96 KB

  packW<<<(NT * HT * 32 * 16 + 255) / 256, 256, 0, stream>>>(W, Wp);
  emitGemm<<<(Bb * Ls) / 16, 128, 0, stream>>>(F, Wp, bias, emit);
  crfScan<<<Bb, 64, 0, stream>>>(emit, trans, tags, mask, out);
}
